// TinyGRU_13623636263553
// MI455X (gfx1250) — matrix operand order repaired and checked by reading, not compiled, not run on hardware
//
#include <hip/hip_runtime.h>
#include <hip/hip_bf16.h>

// ---------------------------------------------------------------------------
// 2-layer GRU, B=32, S=2048, I=H=512, G=1536  (gfx1250 / CDNA5, wave32)
// Persistent kernel, bf16 WMMA, LDS-fused gates, layers software-pipelined:
// tick k runs layer-0 step k AND layer-1 step k-1 concurrently, so only ONE
// grid barrier per tick (2049 total).
//
// Block b owns output columns [b*16, b*16+16). Waves 0..5 compute the six
// layer-0 strips (gi_r/z/n, gh_r/z/n), waves 6..11 the six layer-1 strips,
// all into LDS; each half-block then applies its layer's GRU gates locally.
// ---------------------------------------------------------------------------

#define B_   32
#define S_   2048
#define I_   512
#define H_   512
#define G_   1536        // 3*H
#define NT_  96          // N tiles of 16  (1536/16)
#define KT_  16          // K tiles of 32  (512/32)
#define NBLK 32
#define TPB  384         // 12 waves
#define GTPB 192         // threads per gate group (half block)
#define NTHR (NBLK * TPB)   // 12288

typedef __attribute__((ext_vector_type(16))) __bf16 v16bf;
typedef __attribute__((ext_vector_type(8)))  float  v8f;
typedef __attribute__((ext_vector_type(4)))  unsigned u32x4;
typedef __attribute__((ext_vector_type(4)))  float  f32x4;

union Frag {                 // 32 bytes: one WMMA A or B operand per lane
    v16bf          v;
    u32x4          q[2];
    unsigned       p[8];
    unsigned short u[16];
};

// workspace layout (bytes)
static constexpr size_t HSZF = (size_t)B_ * H_ * 4;    // 65536
static constexpr size_t HSZB = (size_t)B_ * H_ * 2;    // 32768
static constexpr size_t OFF_CNT   = 0;
static constexpr size_t OFF_GEN   = 64;
static constexpr size_t OFF_H0F   = 1024;
static constexpr size_t OFF_H1F   = OFF_H0F + HSZF;
static constexpr size_t OFF_H0B   = OFF_H1F + HSZF;            // 2 parity buffers
static constexpr size_t OFF_H1B   = OFF_H0B + 2 * HSZB;        // 2 parity buffers
static constexpr size_t ZERO_BYTES= OFF_H1B + 2 * HSZB;        // 263168
static constexpr size_t OFF_WPK   = ZERO_BYTES;
static constexpr size_t WPK_ELEMS = (size_t)I_ * G_;           // 786432 bf16 per matrix

__device__ __forceinline__ unsigned short f2bf(float f) {
    unsigned u = __float_as_uint(f);
    unsigned r = u + 0x7FFFu + ((u >> 16) & 1u);   // round-to-nearest-even
    return (unsigned short)(r >> 16);
}
__device__ __forceinline__ unsigned pk2(float lo, float hi) {
    return (unsigned)f2bf(lo) | ((unsigned)f2bf(hi) << 16);
}

// ---------------------------------------------------------------------------
// Weight pack: fp32 [512,1536] row-major -> bf16 B-fragment layout.
// Fragment (nt,kt), lane L, element j:  W[kt*32 + 8*(L/16) + j + (j>=8 ? 8 : 0)][nt*16 + L%16]
// -> each lane's 16 bf16 (32B) contiguous: two global_load_b128 at use.
// ---------------------------------------------------------------------------
__global__ void pack_weights_kernel(const float* __restrict__ w,
                                    unsigned short* __restrict__ dst) {
    int t = blockIdx.x * blockDim.x + threadIdx.x;          // 0 .. 49151
    if (t >= NT_ * KT_ * 32) return;
    int lane = t & 31;
    int kt   = (t >> 5) & 15;
    int nt   = t >> 9;
    int n     = nt * 16 + (lane & 15);
    int kbase = kt * 32 + 8 * (lane >> 4);
    unsigned short* d = dst + (size_t)t * 16;
#pragma unroll
    for (int j = 0; j < 16; ++j)
        d[j] = f2bf(w[(size_t)(kbase + j + (j >= 8 ? 8 : 0)) * G_ + n]);
}

// ---------------------------------------------------------------------------
// grid-wide two-phase barrier (counter + generation, in ws)
// ---------------------------------------------------------------------------
__device__ __forceinline__ void grid_barrier(unsigned* cnt, volatile unsigned* gen) {
    __threadfence();           // release this block's global stores
    __syncthreads();
    if (threadIdx.x == 0) {
        unsigned g = *gen;
        if (atomicAdd(cnt, 1u) == NBLK - 1) {
            *cnt = 0;
            __threadfence();
            atomicAdd((unsigned*)gen, 1u);
        } else {
            while (*gen == g) { __builtin_amdgcn_s_sleep(1); }
        }
    }
    __syncthreads();
    __threadfence();           // acquire: drop stale WGP-cached lines
}

// ---------------------------------------------------------------------------
// One 16-column strip of [32,512]x[512,1536]: A bf16 row-major, result -> LDS
// strip [32 rows][16 cols] f32. Per wave: 2 M-tiles, 16 k-tiles, 32 WMMAs.
// ---------------------------------------------------------------------------
__device__ __forceinline__ void gemm_strip_bf16(const unsigned short* __restrict__ A,
                                                const unsigned short* __restrict__ Wp,
                                                float* __restrict__ lds_out,
                                                int nt, int lane) {
    const int m = lane & 15, half = lane >> 4;
    const unsigned short* a0r = A + (size_t)m * I_;
    const unsigned short* a1r = A + (size_t)(m + 16) * I_;
    const unsigned short* wp  = Wp + ((size_t)nt * KT_ * 32 + lane) * 16;
    v8f acc0 = {}, acc1 = {};
#pragma unroll 4
    for (int kt = 0; kt < KT_; ++kt) {
        Frag bf, a0, a1;
        bf.q[0] = *(const u32x4*)(wp);
        bf.q[1] = *(const u32x4*)(wp + 8);
        __builtin_prefetch(wp + 512, 0, 1);
        const int k1 = kt * 32 + 8 * half;
        a0.q[0] = *(const u32x4*)(a0r + k1);
        a0.q[1] = *(const u32x4*)(a0r + k1 + 16);
        a1.q[0] = *(const u32x4*)(a1r + k1);
        a1.q[1] = *(const u32x4*)(a1r + k1 + 16);
        acc0 = __builtin_amdgcn_wmma_f32_16x16x32_bf16(false, a0.v, false, bf.v,
                                                       (short)0, acc0, false, false);
        acc1 = __builtin_amdgcn_wmma_f32_16x16x32_bf16(false, a1.v, false, bf.v,
                                                       (short)0, acc1, false, false);
        wp += 512;                       // next k-tile fragment (32 lanes * 16)
    }
    float* o = lds_out + (lane & 15);
    const int mrow = 8 * half;
#pragma unroll
    for (int r = 0; r < 8; ++r) {
        o[(mrow + r) * 16]      = acc0[r];
        o[(mrow + r + 16) * 16] = acc1[r];
    }
}

// Same, but A is fp32 (x slice, row stride S*I), converted to bf16 in-register.
__device__ __forceinline__ void gemm_strip_f32(const float* __restrict__ A, size_t rowStride,
                                               const unsigned short* __restrict__ Wp,
                                               float* __restrict__ lds_out,
                                               int nt, int lane) {
    const int m = lane & 15, half = lane >> 4;
    const float* r0 = A + (size_t)m * rowStride;
    const float* r1 = A + (size_t)(m + 16) * rowStride;
    const unsigned short* wp = Wp + ((size_t)nt * KT_ * 32 + lane) * 16;
    v8f acc0 = {}, acc1 = {};
#pragma unroll 4
    for (int kt = 0; kt < KT_; ++kt) {
        Frag bf, a0, a1;
        bf.q[0] = *(const u32x4*)(wp);
        bf.q[1] = *(const u32x4*)(wp + 8);
        const int k1 = kt * 32 + 8 * half;
        f32x4 f0 = *(const f32x4*)(r0 + k1);
        f32x4 f1 = *(const f32x4*)(r0 + k1 + 4);
        f32x4 f2 = *(const f32x4*)(r0 + k1 + 16);
        f32x4 f3 = *(const f32x4*)(r0 + k1 + 20);
        a0.p[0] = pk2(f0[0], f0[1]); a0.p[1] = pk2(f0[2], f0[3]);
        a0.p[2] = pk2(f1[0], f1[1]); a0.p[3] = pk2(f1[2], f1[3]);
        a0.p[4] = pk2(f2[0], f2[1]); a0.p[5] = pk2(f2[2], f2[3]);
        a0.p[6] = pk2(f3[0], f3[1]); a0.p[7] = pk2(f3[2], f3[3]);
        f0 = *(const f32x4*)(r1 + k1);
        f1 = *(const f32x4*)(r1 + k1 + 4);
        f2 = *(const f32x4*)(r1 + k1 + 16);
        f3 = *(const f32x4*)(r1 + k1 + 20);
        a1.p[0] = pk2(f0[0], f0[1]); a1.p[1] = pk2(f0[2], f0[3]);
        a1.p[2] = pk2(f1[0], f1[1]); a1.p[3] = pk2(f1[2], f1[3]);
        a1.p[4] = pk2(f2[0], f2[1]); a1.p[5] = pk2(f2[2], f2[3]);
        a1.p[6] = pk2(f3[0], f3[1]); a1.p[7] = pk2(f3[2], f3[3]);
        acc0 = __builtin_amdgcn_wmma_f32_16x16x32_bf16(false, a0.v, false, bf.v,
                                                       (short)0, acc0, false, false);
        acc1 = __builtin_amdgcn_wmma_f32_16x16x32_bf16(false, a1.v, false, bf.v,
                                                       (short)0, acc1, false, false);
        wp += 512;
    }
    float* o = lds_out + (lane & 15);
    const int mrow = 8 * half;
#pragma unroll
    for (int r = 0; r < 8; ++r) {
        o[(mrow + r) * 16]      = acc0[r];
        o[(mrow + r + 16) * 16] = acc1[r];
    }
}

// Half-block GRU gates over this block's 16 columns x 32 batch rows.
// lds slots: 0..2 = gi_{r,z,n}, 3..5 = gh_{r,z,n}; each [32][16] f32.
__device__ __forceinline__ void gate_phase(const float* __restrict__ lds,
                                           const float* __restrict__ bi,
                                           const float* __restrict__ bh,
                                           float* __restrict__ hf,
                                           unsigned short* __restrict__ hb,
                                           float* __restrict__ outp,  // nullable, = out + t*H
                                           int col0, int e0) {
    for (int e = e0; e < B_ * 16; e += GTPB) {
        int b = e >> 4, n = e & 15;
        int c = col0 + n;                       // column in [0, 512)
        float ir = lds[0 * 512 + e] + bi[c];
        float iz = lds[1 * 512 + e] + bi[c + H_];
        float in = lds[2 * 512 + e] + bi[c + 2 * H_];
        float hr = lds[3 * 512 + e] + bh[c];
        float hz = lds[4 * 512 + e] + bh[c + H_];
        float hn = lds[5 * 512 + e] + bh[c + 2 * H_];
        float r = 1.f / (1.f + __expf(-(ir + hr)));
        float z = 1.f / (1.f + __expf(-(iz + hz)));
        float ng = tanhf(in + r * hn);
        int idx = b * H_ + c;
        float hp = hf[idx];
        float hnew = (1.f - z) * ng + z * hp;
        hf[idx] = hnew;
        hb[idx] = f2bf(hnew);
        if (outp) outp[(size_t)b * S_ * H_ + c] = hnew;
    }
}

// ---------------------------------------------------------------------------
// Persistent recurrent kernel: tick k = { layer-0 step k  ||  layer-1 step k-1 },
// 12 WMMA strips -> LDS, half-block gates, ONE grid barrier per tick.
// ---------------------------------------------------------------------------
__global__ __launch_bounds__(TPB, 1)
void gru_persistent_kernel(const float* __restrict__ x,
                           const float* __restrict__ bih0, const float* __restrict__ bhh0,
                           const float* __restrict__ bih1, const float* __restrict__ bhh1,
                           float* __restrict__ out, char* __restrict__ ws) {
    __shared__ float lds[12 * B_ * 16];         // 24 KB

    unsigned*          cnt = (unsigned*)(ws + OFF_CNT);
    volatile unsigned* gen = (volatile unsigned*)(ws + OFF_GEN);
    float*          h0f = (float*)(ws + OFF_H0F);
    float*          h1f = (float*)(ws + OFF_H1F);
    unsigned short* h0b = (unsigned short*)(ws + OFF_H0B);   // [2][B][H]
    unsigned short* h1b = (unsigned short*)(ws + OFF_H1B);   // [2][B][H]
    const unsigned short* wpk  = (const unsigned short*)(ws + OFF_WPK);
    const unsigned short* wih0 = wpk;
    const unsigned short* whh0 = wpk + WPK_ELEMS;
    const unsigned short* wih1 = wpk + 2 * WPK_ELEMS;
    const unsigned short* whh1 = wpk + 3 * WPK_ELEMS;

    const int w     = (int)(threadIdx.x >> 5);       // wave in block: 0..11
    const int lane  = (int)(threadIdx.x & 31);
    const int layer = w / 6;                         // 0 or 1
    const int ww    = w % 6;
    const int g     = ww % 3;                        // gate chunk r/z/n
    const int side  = ww / 3;                        // 0 = gi, 1 = gh
    const int cg    = (int)blockIdx.x;               // column group 0..31
    const int nt    = 32 * g + cg;                   // N-tile in packed weights
    const int col0  = cg * 16;
    const int gtid  = (int)(blockIdx.x * TPB + threadIdx.x);
    float* ldsS = &lds[(layer * 6 + 3 * side + g) * (B_ * 16)];

#pragma unroll 1
    for (int k = 0; k <= S_; ++k) {                  // 2049 ticks
        const int p = k & 1, q = p ^ 1;              // p: written this tick
        unsigned short* h0_new = h0b + p * (B_ * H_);
        unsigned short* h1_new = h1b + p * (B_ * H_);
        const unsigned short* h0_old = h0b + q * (B_ * H_);   // h0(k-1)
        const unsigned short* h1_old = h1b + q * (B_ * H_);   // h1(k-2)

        // ---- GEMM strips (both layers concurrently) -> LDS ----
        if (layer == 0) {
            if (k < S_) {                            // layer-0 step k
                if (side == 0)
                    gemm_strip_f32(x + (size_t)k * I_, (size_t)S_ * I_, wih0, ldsS, nt, lane);
                else
                    gemm_strip_bf16(h0_old, whh0, ldsS, nt, lane);
            }
        } else if (k >= 1) {                         // layer-1 step k-1
            if (side == 0)
                gemm_strip_bf16(h0_old, wih1, ldsS, nt, lane);
            else
                gemm_strip_bf16(h1_old, whh1, ldsS, nt, lane);
        }
        __syncthreads();

        // ---- gates: half-block per layer ----
        if (layer == 0) {
            if (k < S_)
                gate_phase(lds, bih0, bhh0, h0f, h0_new, nullptr,
                           col0, (int)threadIdx.x);
        } else if (k >= 1) {
            gate_phase(lds + 6 * (B_ * 16), bih1, bhh1, h1f, h1_new,
                       out + (size_t)(k - 1) * H_, col0, (int)threadIdx.x - GTPB);
        }
        grid_barrier(cnt, gen);                      // publish h0(k), h1(k-1)
    }

    // final_hidden [2, B, H] appended after output [B, S, H]
    float* fh = out + (size_t)B_ * S_ * H_;
    for (int idx = gtid; idx < 2 * B_ * H_; idx += NTHR)
        fh[idx] = (idx < B_ * H_) ? h0f[idx] : h1f[idx - B_ * H_];
}

// ---------------------------------------------------------------------------
extern "C" void kernel_launch(void* const* d_in, const int* in_sizes, int n_in,
                              void* d_out, int out_size, void* d_ws, size_t ws_size,
                              hipStream_t stream) {
    (void)in_sizes; (void)n_in; (void)out_size; (void)ws_size;
    const float* x    = (const float*)d_in[0];
    const float* wih0 = (const float*)d_in[1];
    const float* whh0 = (const float*)d_in[2];
    const float* bih0 = (const float*)d_in[3];
    const float* bhh0 = (const float*)d_in[4];
    const float* wih1 = (const float*)d_in[5];
    const float* whh1 = (const float*)d_in[6];
    const float* bih1 = (const float*)d_in[7];
    const float* bhh1 = (const float*)d_in[8];
    float* out = (float*)d_out;
    char*  ws  = (char*)d_ws;

    // zero barrier state + hidden states (deterministic per launch / replay)
    hipMemsetAsync(ws, 0, ZERO_BYTES, stream);

    unsigned short* wpk = (unsigned short*)(ws + OFF_WPK);
    pack_weights_kernel<<<192, 256, 0, stream>>>(wih0, wpk + 0 * WPK_ELEMS);
    pack_weights_kernel<<<192, 256, 0, stream>>>(whh0, wpk + 1 * WPK_ELEMS);
    pack_weights_kernel<<<192, 256, 0, stream>>>(wih1, wpk + 2 * WPK_ELEMS);
    pack_weights_kernel<<<192, 256, 0, stream>>>(whh1, wpk + 3 * WPK_ELEMS);

    gru_persistent_kernel<<<NBLK, TPB, 0, stream>>>(x, bih0, bhh0, bih1, bhh1, out, ws);
}
